// GraphConvSAGE_60413009985910
// MI455X (gfx1250) — compile-verified, matrix operand order checked by reading, not run on hardware
//
#include <hip/hip_runtime.h>
#include <hip/hip_bf16.h>

#define NN 50000
#define NE 800000
#define NG 64
#define D 128
#define DOUT 32
#define GEMM_CHUNK 4   // row tiles per wave (B operand held in registers)

typedef __attribute__((ext_vector_type(16))) __bf16 v16bf;
typedef __attribute__((ext_vector_type(8)))  float  v8f;

__device__ __forceinline__ float bf2f(unsigned short u) {
  unsigned int x = ((unsigned int)u) << 16;
  return __builtin_bit_cast(float, x);
}
__device__ __forceinline__ unsigned short f2bf(float f) {
  unsigned int x = __builtin_bit_cast(unsigned int, f);
  x += 0x7fffu + ((x >> 16) & 1u);   // round-to-nearest-even
  return (unsigned short)(x >> 16);
}
__device__ __forceinline__ void atomAddF(float* p, float v) { unsafeAtomicAdd(p, v); }

// ---------------- utility ----------------
__global__ void k_zero(float* __restrict__ p, int n) {
  int i = blockIdx.x * blockDim.x + threadIdx.x;
  if (i < n) p[i] = 0.0f;
}

// ---------------- feature standardization ----------------
__global__ void k_stats(const float* __restrict__ x, float* __restrict__ sums, int n_nodes) {
  int f = threadIdx.x;                 // 128 threads = 128 features
  int r0 = blockIdx.x * 256;
  int r1 = r0 + 256 < n_nodes ? r0 + 256 : n_nodes;
  float s = 0.f, sq = 0.f;
  for (int r = r0; r < r1; ++r) {
    float v = x[r * D + f];
    s += v; sq += v * v;
  }
  atomAddF(&sums[f], s);
  atomAddF(&sums[D + f], sq);
}

__global__ void k_stats_fin(const float* __restrict__ sums, float* __restrict__ muinv) {
  int f = threadIdx.x;
  float mu  = sums[f] * (1.0f / NN);
  float var = sums[D + f] * (1.0f / NN) - mu * mu;
  var = fmaxf(var, 0.0f);
  muinv[f]     = mu;
  muinv[D + f] = 1.0f / sqrtf(var);
}

__global__ void k_normalize(const float* __restrict__ x, const float* __restrict__ muinv,
                            unsigned short* __restrict__ hb, int total) {
  int i = blockIdx.x * blockDim.x + threadIdx.x;
  if (i >= total) return;
  int f = i & (D - 1);
  hb[i] = f2bf((x[i] - muinv[f]) * muinv[D + f]);
}

// ---------------- degree ----------------
__global__ void k_degree(const int* __restrict__ ei, float* __restrict__ deg, int E) {
  int e = blockIdx.x * blockDim.x + threadIdx.x;
  if (e < E) atomAddF(&deg[ei[E + e]], 1.0f);
}
__global__ void k_invdeg(const float* __restrict__ deg, float* __restrict__ inv, int n) {
  int i = blockIdx.x * blockDim.x + threadIdx.x;
  if (i < n) inv[i] = 1.0f / fmaxf(deg[i], 1.0f);
}

// ---------------- edge scatter: agg[dst] += h[src] (bf16 gather, f32 atomic add) ----
// One wave per edge; edge id forced wave-uniform so src/dst become scalar loads.
__global__ void k_scatter(const unsigned short* __restrict__ h, const int* __restrict__ ei,
                          float* __restrict__ agg, int E) {
  int gid = blockIdx.x * blockDim.x + threadIdx.x;
  int e = __builtin_amdgcn_readfirstlane(gid >> 5);
  if (e >= E) return;
  int lane = gid & 31;
  int src = ei[e];
  int dst = ei[E + e];
  const unsigned short* hp = h + src * D + lane * 4;
  uint2 pk = *(const uint2*)hp;        // 4 bf16 values
  float* ap = agg + dst * D + lane * 4;
  atomAddF(ap + 0, bf2f((unsigned short)(pk.x & 0xffffu)));
  atomAddF(ap + 1, bf2f((unsigned short)(pk.x >> 16)));
  atomAddF(ap + 2, bf2f((unsigned short)(pk.y & 0xffffu)));
  atomAddF(ap + 3, bf2f((unsigned short)(pk.y >> 16)));
}

// ---------------- agg finalize: (optional /deg) -> bf16 ----------------
__global__ void k_aggfin(const float* __restrict__ agg, const float* __restrict__ invdeg,
                         int use_deg, unsigned short* __restrict__ aggb, int total) {
  int i = blockIdx.x * blockDim.x + threadIdx.x;
  if (i >= total) return;
  float v = agg[i];
  if (use_deg) v *= invdeg[i >> 7];
  aggb[i] = f2bf(v);
}

// ---------------- weights: f32 [k][n] -> bf16 transposed [n][k] ----------------
__global__ void k_wconv(const float* __restrict__ w, unsigned short* __restrict__ wT) {
  int i = blockIdx.x * blockDim.x + threadIdx.x;   // 128*128
  int k = i >> 7, n = i & (D - 1);
  wT[n * D + k] = f2bf(w[i]);
}

// ---------------- fused dual-GEMM: out = act(A0*W0 + bias + A1*W1) ----------------
// Wave = one 16-column slice (tn = wave id), B operands for K=128 held in registers
// (4 k-steps x 8 VGPRs x 2 matrices), loop over GEMM_CHUNK row tiles reusing B.
// The 8 waves of a block cover all 8 column tiles of the same rows -> shared A.
__global__ __launch_bounds__(256) void k_gemm(
    const unsigned short* __restrict__ A0, const unsigned short* __restrict__ W0T,
    const unsigned short* __restrict__ A1, const unsigned short* __restrict__ W1T,
    const float* __restrict__ bias, float* __restrict__ outF,
    unsigned short* __restrict__ outB, int n_tiles_m, int do_relu) {
  int tn   = threadIdx.x >> 5;         // 8 waves = 8 column tiles (128/16)
  int lane = threadIdx.x & 31;
  int tmBase = blockIdx.x * GEMM_CHUNK;
  if (tmBase >= n_tiles_m) return;
  int half = lane >> 4;
  int mn = lane & 15;
  int n0 = tn << 4;

  union U { unsigned int u[8]; v16bf v; };

  // Preload both weight operands for this column slice, all of K.
  U b0[4], b1[4];
  const unsigned short* br0 = W0T + (n0 + mn) * D;
  const unsigned short* br1 = W1T + (n0 + mn) * D;
#pragma unroll
  for (int ks = 0; ks < 4; ++ks) {
#pragma unroll
    for (int v = 0; v < 8; ++v) {
      int k = (ks << 5) + ((v & 3) << 1) + (half << 3) + ((v >> 2) << 4);
      b0[ks].u[v] = *(const unsigned int*)(br0 + k);
      b1[ks].u[v] = *(const unsigned int*)(br1 + k);
    }
  }
  float bv = bias[n0 + mn];

  int tmEnd = tmBase + GEMM_CHUNK;
  if (tmEnd > n_tiles_m) tmEnd = n_tiles_m;
  for (int tm = tmBase; tm < tmEnd; ++tm) {
    int m0 = tm << 4;
    v8f c;
#pragma unroll
    for (int v = 0; v < 8; ++v) c[v] = bv;

    const unsigned short* ar0 = A0 + (m0 + mn) * D;
#pragma unroll
    for (int ks = 0; ks < 4; ++ks) {
      U a;
#pragma unroll
      for (int v = 0; v < 8; ++v) {
        int k = (ks << 5) + ((v & 3) << 1) + (half << 3) + ((v >> 2) << 4);
        a.u[v] = *(const unsigned int*)(ar0 + k);
      }
      c = __builtin_amdgcn_wmma_f32_16x16x32_bf16(false, a.v, false, b0[ks].v,
                                                  (short)0, c, false, false);
    }
    const unsigned short* ar1 = A1 + (m0 + mn) * D;
#pragma unroll
    for (int ks = 0; ks < 4; ++ks) {
      U a;
#pragma unroll
      for (int v = 0; v < 8; ++v) {
        int k = (ks << 5) + ((v & 3) << 1) + (half << 3) + ((v >> 2) << 4);
        a.u[v] = *(const unsigned int*)(ar1 + k);
      }
      c = __builtin_amdgcn_wmma_f32_16x16x32_bf16(false, a.v, false, b1[ks].v,
                                                  (short)0, c, false, false);
    }

    if (do_relu) {
#pragma unroll
      for (int v = 0; v < 8; ++v) c[v] = fmaxf(c[v], 0.0f);
    }
#pragma unroll
    for (int v = 0; v < 8; ++v) {
      int idx = (m0 + v + (half << 3)) * D + n0 + mn;   // C/D layout: VGPR v -> M=v(+8), N=lane
      outF[idx] = c[v];
      outB[idx] = f2bf(c[v]);
    }
  }
}

// ---------------- pooling + FC head ----------------
__global__ void k_pool(const float* __restrict__ h, const int* __restrict__ batch,
                       float* __restrict__ pooled, float* __restrict__ cnt, int total) {
  int i = blockIdx.x * blockDim.x + threadIdx.x;
  if (i >= total) return;
  int n = i >> 7, f = i & (D - 1);
  int g = batch[n];
  atomAddF(&pooled[g * D + f], h[i]);
  if (f == 0) atomAddF(&cnt[g], 1.0f);
}

__global__ void k_fc(const float* __restrict__ pooled, const float* __restrict__ cnt,
                     const float* __restrict__ w, const float* __restrict__ b,
                     float* __restrict__ out) {
  int i = blockIdx.x * blockDim.x + threadIdx.x;
  if (i >= NG * DOUT) return;
  int g = i >> 5, o = i & (DOUT - 1);
  float inv = 1.0f / fmaxf(cnt[g], 1.0f);
  float acc = b[o];
  for (int k = 0; k < D; ++k) acc += pooled[g * D + k] * inv * w[k * DOUT + o];
  out[g * DOUT + o] = acc;
}

// ---------------- host side ----------------
static inline size_t alignup(size_t v) { return (v + 255) & ~size_t(255); }

extern "C" void kernel_launch(void* const* d_in, const int* in_sizes, int n_in,
                              void* d_out, int out_size, void* d_ws, size_t ws_size,
                              hipStream_t stream) {
  (void)in_sizes; (void)n_in; (void)out_size; (void)ws_size;
  const float* x      = (const float*)d_in[0];
  const int*   ei     = (const int*)d_in[1];
  const int*   batch  = (const int*)d_in[2];
  const float* w_rel  = (const float*)d_in[3];
  const float* b_rel  = (const float*)d_in[4];
  const float* w_root = (const float*)d_in[5];
  const float* sw_l   = (const float*)d_in[6];
  const float* sb_l   = (const float*)d_in[7];
  const float* sw_r   = (const float*)d_in[8];
  const float* fc_w   = (const float*)d_in[9];
  const float* fc_b   = (const float*)d_in[10];
  float* out = (float*)d_out;

  char* ws = (char*)d_ws;
  size_t off = 0;
  auto alloc = [&](size_t bytes) { char* p = ws + off; off += alignup(bytes); return p; };

  unsigned short* hb0  = (unsigned short*)alloc((size_t)NN * D * 2);
  unsigned short* hb1  = (unsigned short*)alloc((size_t)NN * D * 2);
  float*          aggf = (float*)alloc((size_t)NN * D * 4);
  unsigned short* aggb = (unsigned short*)alloc((size_t)NN * D * 2);
  float*          hf   = (float*)alloc((size_t)NN * D * 4);
  float*          deg  = (float*)alloc((size_t)NN * 4);
  float*          invd = (float*)alloc((size_t)NN * 4);
  unsigned short* wT0  = (unsigned short*)alloc((size_t)D * D * 2);
  unsigned short* wT1  = (unsigned short*)alloc((size_t)D * D * 2);
  float*          stats  = (float*)alloc(2 * D * 4);
  float*          muinv  = (float*)alloc(2 * D * 4);
  float*          pooled = (float*)alloc((size_t)NG * D * 4);
  float*          cnt    = (float*)alloc((size_t)NG * 4);

  const int total = NN * D;
  const int B = 256;

  // 1. standardization
  k_zero<<<1, 256, 0, stream>>>(stats, 2 * D);
  k_stats<<<(NN + 255) / 256, D, 0, stream>>>(x, stats, NN);
  k_stats_fin<<<1, D, 0, stream>>>(stats, muinv);
  k_normalize<<<(total + B - 1) / B, B, 0, stream>>>(x, muinv, hb0, total);

  // 2. degree
  k_zero<<<(NN + B - 1) / B, B, 0, stream>>>(deg, NN);
  k_degree<<<(NE + B - 1) / B, B, 0, stream>>>(ei, deg, NE);
  k_invdeg<<<(NN + B - 1) / B, B, 0, stream>>>(deg, invd, NN);

  // 3. layers
  const float* WA[4] = { w_rel, w_rel + D * D, sw_l, sw_l + D * D };        // neighbor weights
  const float* WB[4] = { w_root, w_root + D * D, sw_r, sw_r + D * D };      // root weights
  const float* BI[4] = { b_rel, b_rel + D, sb_l, sb_l + D };
  const int useDeg[4] = { 0, 0, 1, 1 };
  const int doRelu[4] = { 1, 1, 1, 0 };

  unsigned short* cur = hb0;
  unsigned short* nxt = hb1;
  const int tilesM = NN / 16;                                   // 3125, exact
  const int gemmBlocks = (tilesM + GEMM_CHUNK - 1) / GEMM_CHUNK; // 782

  for (int l = 0; l < 4; ++l) {
    k_wconv<<<(D * D + B - 1) / B, B, 0, stream>>>(WA[l], wT0);
    k_wconv<<<(D * D + B - 1) / B, B, 0, stream>>>(WB[l], wT1);
    k_zero<<<(total + B - 1) / B, B, 0, stream>>>(aggf, total);
    k_scatter<<<((size_t)NE * 32 + B - 1) / B, B, 0, stream>>>(cur, ei, aggf, NE);
    k_aggfin<<<(total + B - 1) / B, B, 0, stream>>>(aggf, invd, useDeg[l], aggb, total);
    k_gemm<<<gemmBlocks, 256, 0, stream>>>(aggb, wT0, cur, wT1, BI[l], hf, nxt,
                                           tilesM, doRelu[l]);
    unsigned short* t = cur; cur = nxt; nxt = t;
  }

  // 4. pool + FC
  k_zero<<<(NG * D + B - 1) / B, B, 0, stream>>>(pooled, NG * D);
  k_zero<<<1, NG, 0, stream>>>(cnt, NG);
  k_pool<<<(total + B - 1) / B, B, 0, stream>>>(hf, batch, pooled, cnt, total);
  k_fc<<<(NG * DOUT + B - 1) / B, B, 0, stream>>>(pooled, cnt, fc_w, fc_b, out);
}
